// InfoNCE_27092653703229
// MI455X (gfx1250) — matrix operand order repaired and checked by reading, not compiled, not run on hardware
//
#include <hip/hip_runtime.h>

// ---------------------------------------------------------------------------
// InfoNCE lower bound on MI455X (gfx1250, wave32, WMMA).
//   S[i,b] = x[i] @ W @ y[b];  out = mean(diag S) - (mean_b lse_i S[i,b] - log N)
// Big GEMM (8192x8192x256 = 34.4 GFLOP) is compute bound -> bf16 WMMA with
// f32 accumulate, logsumexp fused into the epilogue (S never materialized).
// Operands staged in LDS (padded, bank-rotated) via async global->LDS copies
// (GLOBAL_LOAD_ASYNC_TO_LDS_B128, drained with s_wait_asynccnt).
// ---------------------------------------------------------------------------

#define N_SAMP   8192
#define DIM      256
#define ROWCHUNK 256                   // rows per block in the LSE GEMM
#define NCHUNKS  (N_SAMP / ROWCHUNK)   // 32
#define LDA      264                   // padded LDS row stride (bf16 elems)

typedef __attribute__((ext_vector_type(16))) __bf16 v16bf;
typedef __attribute__((ext_vector_type(8)))  __bf16 v8bf;
typedef __attribute__((ext_vector_type(8)))  float  v8f;
typedef int v4i __attribute__((vector_size(16)));

union FragU { v16bf v; v8bf h[2]; };

#if defined(__gfx1250__) && \
    __has_builtin(__builtin_amdgcn_global_load_async_to_lds_b128) && \
    __has_builtin(__builtin_amdgcn_s_wait_asynccnt)
#define HAVE_ASYNC_LDS 1
#else
#define HAVE_ASYNC_LDS 0
#endif

// Copy 16 bytes global -> LDS. Async DMA (ASYNCcnt) when available, else a
// plain load + ds_store (still stages through LDS).
__device__ __forceinline__ void copy16_g2l(__bf16* lds_dst, const __bf16* gsrc) {
#if HAVE_ASYNC_LDS
    __builtin_amdgcn_global_load_async_to_lds_b128(
        (__attribute__((address_space(1))) v4i*)(void*)const_cast<__bf16*>(gsrc),
        (__attribute__((address_space(3))) v4i*)(void*)lds_dst,
        0, 0);
#else
    *(v8bf*)lds_dst = *(const v8bf*)gsrc;
#endif
}

__device__ __forceinline__ void g2l_fence() {
#if HAVE_ASYNC_LDS
    __builtin_amdgcn_s_wait_asynccnt(0);
#endif
    __syncthreads();
}

// A-matrix 16x32 bf16 fragment (ISA 7.12.2): lane L holds row (L&15),
// K chunks at k0+(L>>4)*8 and k0+16+(L>>4)*8 (each 8 bf16 = 16B).
__device__ __forceinline__ v16bf load_a_frag(const __bf16* base, int ld,
                                             int row0, int k0, int lane) {
    const __bf16* p = base + (size_t)(row0 + (lane & 15)) * ld
                           + k0 + ((lane >> 4) << 3);
    FragU f;
    f.h[0] = *(const v8bf*)(p);
    f.h[1] = *(const v8bf*)(p + 16);
    return f.v;
}

// B-matrix 32x16 bf16 fragment: lane L holds column (L&15), K chunks at
// k0+(L>>4)*8 and k0+16+(L>>4)*8 -> rows of a row-major [N,K] operand.
__device__ __forceinline__ v16bf load_b_frag(const __bf16* base, int ld,
                                             int col0, int k0, int lane) {
    const __bf16* p = base + (size_t)(col0 + (lane & 15)) * ld
                           + k0 + ((lane >> 4) << 3);
    FragU f;
    f.h[0] = *(const v8bf*)(p);
    f.h[1] = *(const v8bf*)(p + 16);
    return f.v;
}

__device__ __forceinline__ v8f wmma_bf16(v16bf a, v16bf b, v8f c) {
    return __builtin_amdgcn_wmma_f32_16x16x32_bf16(
        /*neg_a=*/false, a, /*neg_b=*/false, b,
        /*c_mod=*/(short)0, c, /*reuse_a=*/false, /*reuse_b=*/false);
}

// Online (max, sum-exp) update with 8 new values (one C-fragment column slice).
__device__ __forceinline__ void online_update(float& rm, float& rs, const v8f& a) {
    float m8 = a[0];
#pragma unroll
    for (int j = 1; j < 8; ++j) m8 = fmaxf(m8, a[j]);
    float nm = fmaxf(rm, m8);
    float s8 = 0.f;
#pragma unroll
    for (int j = 0; j < 8; ++j) s8 += __expf(a[j] - nm);
    rs = rs * __expf(rm - nm) + s8;
    rm = nm;
}

// ---------------------------------------------------------------------------
// Kernel 1: f32 -> bf16 conversions. Also writes W transposed ([y_dim][x_dim])
// so B-fragments of the xw GEMM load contiguously.
// ---------------------------------------------------------------------------
__global__ __launch_bounds__(256) void k_convert(const float* __restrict__ x,
                                                 const float* __restrict__ y,
                                                 const float* __restrict__ w,
                                                 __bf16* __restrict__ xb,
                                                 __bf16* __restrict__ yb,
                                                 __bf16* __restrict__ wtb) {
    size_t t = (size_t)blockIdx.x * blockDim.x + threadIdx.x;
    if (t < (size_t)N_SAMP * DIM) {
        xb[t] = (__bf16)x[t];
        yb[t] = (__bf16)y[t];
    }
    if (t < (size_t)DIM * DIM) {
        int n = (int)(t >> 8);       // row of w^T  (y_dim index)
        int k = (int)(t & 255);      // col of w^T  (x_dim index)
        wtb[t] = (__bf16)w[(size_t)k * DIM + n];
    }
}

// ---------------------------------------------------------------------------
// Kernel 2: xw = x @ W   (8192x256x256, 3% of total FLOPs), result bf16.
// Direct-from-global fragments; grid (DIM/128, N_SAMP/64) = (2, 128).
// ---------------------------------------------------------------------------
__global__ __launch_bounds__(256) void k_gemm_xw(const __bf16* __restrict__ xb,
                                                 const __bf16* __restrict__ wtb,
                                                 __bf16* __restrict__ xwb) {
    const int lane = threadIdx.x & 31;
    const int wid  = threadIdx.x >> 5;
    const int mw = wid & 1;
    const int nw = wid >> 1;
    const int row0 = blockIdx.y * 64 + mw * 32;
    const int col0 = blockIdx.x * 128 + nw * 32;

    v8f acc[2][2] = {};
#pragma unroll
    for (int k0 = 0; k0 < DIM; k0 += 32) {
        v16bf a0 = load_a_frag(xb, DIM, row0,      k0, lane);
        v16bf a1 = load_a_frag(xb, DIM, row0 + 16, k0, lane);
        v16bf b0 = load_b_frag(wtb, DIM, col0,      k0, lane);
        v16bf b1 = load_b_frag(wtb, DIM, col0 + 16, k0, lane);
        acc[0][0] = wmma_bf16(a0, b0, acc[0][0]);
        acc[0][1] = wmma_bf16(a0, b1, acc[0][1]);
        acc[1][0] = wmma_bf16(a1, b0, acc[1][0]);
        acc[1][1] = wmma_bf16(a1, b1, acc[1][1]);
    }
    // C layout: VGPR r, lane L -> row = r + (L>>4)*8, col = (L&15)
#pragma unroll
    for (int mi = 0; mi < 2; ++mi)
#pragma unroll
        for (int ni = 0; ni < 2; ++ni)
#pragma unroll
            for (int r = 0; r < 8; ++r) {
                int row = row0 + mi * 16 + r + ((lane >> 4) << 3);
                int col = col0 + ni * 16 + (lane & 15);
                xwb[(size_t)row * DIM + col] = (__bf16)acc[mi][ni][r];
            }
}

// ---------------------------------------------------------------------------
// Kernel 3: S = xw @ y^T with fused column-wise online LSE, LDS-staged.
// Grid: (N_SAMP/128 column tiles, NCHUNKS row chunks). Block = 8 waves (2x4).
// B tile (128x256 bf16) resident in LDS for the whole block; A slabs (64x256)
// streamed through LDS. Padded stride LDA=264 rotates banks (528B = 132 words,
// 132 % 64 = 4 -> consecutive rows hit disjoint bank quads).
// ---------------------------------------------------------------------------
__global__ __launch_bounds__(256) void k_score_lse(const __bf16* __restrict__ xwb,
                                                   const __bf16* __restrict__ yb,
                                                   float* __restrict__ pmax,
                                                   float* __restrict__ psum) {
    __shared__ __attribute__((aligned(16))) __bf16 sA[64 * LDA];   // 33 KB
    __shared__ __attribute__((aligned(16))) __bf16 sB[128 * LDA];  // 66 KB
    __shared__ float smax[2][128];
    __shared__ float ssum[2][128];

    const int lane = threadIdx.x & 31;
    const int wid  = threadIdx.x >> 5;
    const int mw = wid & 1;           // row-wave within slab
    const int nw = wid >> 1;          // column-wave (0..3)
    const int cbase = blockIdx.x * 128;
    const int col0  = nw * 32;        // column offset inside LDS B tile
    const int rchunk = blockIdx.y;

    // Stage B tile: 128 rows x 256 bf16 = 4096 x 16B chunks (32 chunks/row).
    for (int c = threadIdx.x; c < 128 * 32; c += 256) {
        int row = c >> 5, seg = c & 31;
        copy16_g2l(&sB[row * LDA + seg * 8],
                   yb + (size_t)(cbase + row) * DIM + seg * 8);
    }

    float rm[2] = { -__builtin_inff(), -__builtin_inff() };
    float rs[2] = { 0.f, 0.f };

    for (int it = 0; it < ROWCHUNK / 64; ++it) {
        const int rowbase = rchunk * ROWCHUNK + it * 64;
        // Stage A slab: 64 rows x 256 bf16 = 2048 x 16B chunks.
        for (int c = threadIdx.x; c < 64 * 32; c += 256) {
            int row = c >> 5, seg = c & 31;
            copy16_g2l(&sA[row * LDA + seg * 8],
                       xwb + (size_t)(rowbase + row) * DIM + seg * 8);
        }
        g2l_fence();   // drain ASYNCcnt (or ds stores) + block barrier

        const int arow0 = mw * 32;
        v8f acc[2][2] = {};
#pragma unroll
        for (int k0 = 0; k0 < DIM; k0 += 32) {
            v16bf a0 = load_a_frag(sA, LDA, arow0,      k0, lane);
            v16bf a1 = load_a_frag(sA, LDA, arow0 + 16, k0, lane);
            v16bf b0 = load_b_frag(sB, LDA, col0,       k0, lane);
            v16bf b1 = load_b_frag(sB, LDA, col0 + 16,  k0, lane);
            acc[0][0] = wmma_bf16(a0, b0, acc[0][0]);
            acc[0][1] = wmma_bf16(a0, b1, acc[0][1]);
            acc[1][0] = wmma_bf16(a1, b0, acc[1][0]);
            acc[1][1] = wmma_bf16(a1, b1, acc[1][1]);
        }
        // Each lane's v8f holds 8 rows of a single column -> per-column update.
#pragma unroll
        for (int mi = 0; mi < 2; ++mi) {
            online_update(rm[0], rs[0], acc[mi][0]);
            online_update(rm[1], rs[1], acc[mi][1]);
        }
        __syncthreads();   // protect sA before next slab overwrites it
    }

    // Combine lane pairs (L, L^16): same column, disjoint row sets.
#pragma unroll
    for (int ni = 0; ni < 2; ++ni) {
        float om = __shfl_xor(rm[ni], 16, 32);
        float os = __shfl_xor(rs[ni], 16, 32);
        float nm = fmaxf(rm[ni], om);
        rs[ni] = rs[ni] * __expf(rm[ni] - nm) + os * __expf(om - nm);
        rm[ni] = nm;
    }

    // Stage per-wave results (lanes 0..15 hold columns nw*32+ni*16+lane).
    if (lane < 16) {
#pragma unroll
        for (int ni = 0; ni < 2; ++ni) {
            int cl = nw * 32 + ni * 16 + lane;
            smax[mw][cl] = rm[ni];
            ssum[mw][cl] = rs[ni];
        }
    }
    __syncthreads();

    // Merge the two row-waves and emit one (max,sum) partial per column.
    int t = threadIdx.x;
    if (t < 128) {
        float m0 = smax[0][t], m1 = smax[1][t];
        float s0 = ssum[0][t], s1 = ssum[1][t];
        float m = fmaxf(m0, m1);
        float s = s0 * __expf(m0 - m) + s1 * __expf(m1 - m);
        size_t o = (size_t)rchunk * N_SAMP + cbase + t;
        pmax[o] = m;
        psum[o] = s;
    }
}

// ---------------------------------------------------------------------------
// Kernel 4: diag[i] = xw[i] . y[i]
// ---------------------------------------------------------------------------
__global__ __launch_bounds__(256) void k_diag(const __bf16* __restrict__ xwb,
                                              const __bf16* __restrict__ yb,
                                              float* __restrict__ diag) {
    int i = blockIdx.x * blockDim.x + threadIdx.x;
    if (i >= N_SAMP) return;
    const __bf16* a = xwb + (size_t)i * DIM;
    const __bf16* b = yb  + (size_t)i * DIM;
    float acc = 0.f;
#pragma unroll 8
    for (int k = 0; k < DIM; ++k) acc += (float)a[k] * (float)b[k];
    diag[i] = acc;
}

// ---------------------------------------------------------------------------
// Kernel 5: merge chunk partials -> per-column LSE -> means -> scalar bound.
// ---------------------------------------------------------------------------
__global__ __launch_bounds__(256) void k_finalize(const float* __restrict__ pmax,
                                                  const float* __restrict__ psum,
                                                  const float* __restrict__ diag,
                                                  float* __restrict__ out) {
    __shared__ float red_lse[256];
    __shared__ float red_dia[256];
    int t = threadIdx.x;
    float lse_acc = 0.f, dia_acc = 0.f;
    for (int c = t; c < N_SAMP; c += 256) {
        float m = -__builtin_inff();
        for (int r = 0; r < NCHUNKS; ++r)
            m = fmaxf(m, pmax[(size_t)r * N_SAMP + c]);
        float s = 0.f;
        for (int r = 0; r < NCHUNKS; ++r)
            s += psum[(size_t)r * N_SAMP + c] *
                 __expf(pmax[(size_t)r * N_SAMP + c] - m);
        lse_acc += m + __logf(s);
        dia_acc += diag[c];
    }
    red_lse[t] = lse_acc;
    red_dia[t] = dia_acc;
    __syncthreads();
    for (int w = 128; w > 0; w >>= 1) {
        if (t < w) {
            red_lse[t] += red_lse[t + w];
            red_dia[t] += red_dia[t + w];
        }
        __syncthreads();
    }
    if (t == 0) {
        const float n = (float)N_SAMP;
        float t0_mean  = red_dia[0] / n;
        float lse_mean = red_lse[0] / n;
        out[0] = t0_mean - (lse_mean - __logf(n));
    }
}

// ---------------------------------------------------------------------------
// Host launcher
// ---------------------------------------------------------------------------
extern "C" void kernel_launch(void* const* d_in, const int* in_sizes, int n_in,
                              void* d_out, int out_size, void* d_ws, size_t ws_size,
                              hipStream_t stream) {
    (void)in_sizes; (void)n_in; (void)out_size; (void)ws_size;
    const float* x = (const float*)d_in[0];   // [8192,256]
    const float* y = (const float*)d_in[1];   // [8192,256]
    const float* w = (const float*)d_in[2];   // [256,256]
    float* out = (float*)d_out;

    // Workspace carve-up (256B aligned).
    char* ws = (char*)d_ws;
    size_t off = 0;
    auto carve = [&](size_t bytes) {
        char* p = ws + off;
        off = (off + bytes + 255) & ~(size_t)255;
        return p;
    };
    const size_t n_elems = (size_t)N_SAMP * DIM;
    __bf16* xb   = (__bf16*)carve(n_elems * sizeof(__bf16));   // 4 MB
    __bf16* yb   = (__bf16*)carve(n_elems * sizeof(__bf16));   // 4 MB
    __bf16* xwb  = (__bf16*)carve(n_elems * sizeof(__bf16));   // 4 MB
    __bf16* wtb  = (__bf16*)carve((size_t)DIM * DIM * sizeof(__bf16));
    float*  diag = (float*)carve((size_t)N_SAMP * sizeof(float));
    float*  pmax = (float*)carve((size_t)NCHUNKS * N_SAMP * sizeof(float));
    float*  psum = (float*)carve((size_t)NCHUNKS * N_SAMP * sizeof(float));

    // 1) convert to bf16 (+ transposed W)
    {
        int total = (int)(n_elems);
        k_convert<<<(total + 255) / 256, 256, 0, stream>>>(x, y, w, xb, yb, wtb);
    }
    // 2) xw = x @ W
    {
        dim3 grid(DIM / 128, N_SAMP / 64);   // (2, 128)
        k_gemm_xw<<<grid, 256, 0, stream>>>(xb, wtb, xwb);
    }
    // 3) fused GEMM + column LSE partials (LDS staged)
    {
        dim3 grid(N_SAMP / 128, NCHUNKS);    // (64, 32)
        k_score_lse<<<grid, 256, 0, stream>>>(xwb, yb, pmax, psum);
    }
    // 4) diagonal
    k_diag<<<N_SAMP / 256, 256, 0, stream>>>(xwb, yb, diag);
    // 5) reduce to scalar
    k_finalize<<<1, 256, 0, stream>>>(pmax, psum, diag, out);
}
